// Block_24661702213802
// MI455X (gfx1250) — matrix operand order repaired and checked by reading, not compiled, not run on hardware
//
#include <hip/hip_runtime.h>
#include <hip/hip_bf16.h>
#include <math.h>

typedef __attribute__((ext_vector_type(16))) _Float16 v16h;
typedef __attribute__((ext_vector_type(8)))  _Float16 v8h;
typedef __attribute__((ext_vector_type(8)))  float    v8f;

static __device__ __forceinline__ v16h cat8(v8h lo, v8h hi) {
    return __builtin_shufflevector(lo, hi, 0,1,2,3,4,5,6,7,8,9,10,11,12,13,14,15);
}

static __device__ __forceinline__ v8f wmma16(v16h a, v16h b, v8f c) {
    // D = A(16x32 f16) * B(32x16 f16) + C(16x16 f32)
    return __builtin_amdgcn_wmma_f32_16x16x32_f16(false, a, false, b, (short)0, c, false, false);
}

// ---- A fragment from LDS: row-major tile, stride 40 halves (80B, 16B aligned) ----
static __device__ __forceinline__ v16h lds_afrag(const _Float16* s, int rowbase, int lane) {
    int row = rowbase + (lane & 15);
    int h   = lane >> 4;
    v8h lo = *(const v8h*)(s + row*40 + h*8);
    v8h hi = *(const v8h*)(s + row*40 + 16 + h*8);
    return cat8(lo, hi);
}

// ---- B fragment from LDS: tile stored [N][K], stride 40 halves ----
static __device__ __forceinline__ v16h lds_bfrag(const _Float16* s, int colbase, int lane) {
    int col = colbase + (lane & 15);
    int kh  = (lane >> 4) * 8;
    v8h lo = *(const v8h*)(s + col*40 + kh);
    v8h hi = *(const v8h*)(s + col*40 + kh + 16);
    return cat8(lo, hi);
}

// =====================  elementwise: DyT -> f16  =====================
__global__ __launch_bounds__(256)
void dyt_f16_kernel(const float* __restrict__ x, const float* __restrict__ alpha,
                    const float* __restrict__ gamma, const float* __restrict__ beta,
                    _Float16* __restrict__ h) {
    size_t i = (size_t)blockIdx.x * 256 + threadIdx.x;
    int c = (int)(i & 1023);
    float a = alpha[0];
    h[i] = (_Float16)(gamma[c] * tanhf(x[i] * a) + beta[c]);
}

// ============  weight convert f32 [K,N] -> f16 transposed [N,K]  ============
__global__ __launch_bounds__(256)
void cvt_transpose_kernel(const float* __restrict__ in, _Float16* __restrict__ out,
                          int K, int N) {
    size_t i = (size_t)blockIdx.x * 256 + threadIdx.x;
    int k = (int)(i / (size_t)N);
    int n = (int)(i % (size_t)N);
    out[(size_t)n * K + k] = (_Float16)in[i];
}

// =====================  generic WMMA GEMM, 128x128 block tile  =====================
// 8 waves = 4(M) x 2(N); wave tile 32(M) x 64(N) = 2x4 WMMA accumulators.
// MODE 0: qkv   (bias; scatter Q,K row-major [bh][t][d], V transposed [bh][d][t], all f16)
// MODE 1: proj  (bias + resid x -> x1 f32; also h2 = dyt(x1) f16)
// MODE 2: fc    (bias + exact GELU -> f16)
// MODE 3: fc2   (bias + resid x1 -> f32 out)
template<int MODE>
__global__ __launch_bounds__(256)
void gemm_wmma(const _Float16* __restrict__ A, const _Float16* __restrict__ BT,
               const float* __restrict__ bias, int M, int N, int K,
               const float* __restrict__ resid,
               float* __restrict__ outf, _Float16* __restrict__ outh,
               _Float16* __restrict__ qb, _Float16* __restrict__ kbuf, _Float16* __restrict__ vtb,
               const float* __restrict__ alpha, const float* __restrict__ gamma,
               const float* __restrict__ beta) {
    __shared__ __align__(16) _Float16 sA[128 * 40];
    __shared__ __align__(16) _Float16 sB[128 * 40];

    const int tid  = threadIdx.x;
    const int lane = tid & 31, wid = tid >> 5;
    const int wm = wid & 3, wn = wid >> 2;           // 4 x 2 wave grid
    const int n_ = lane & 15, hf = lane >> 4;
    const int mbase = blockIdx.y * 128;
    const int nbase = blockIdx.x * 128;

    v8f acc[2][4] = {};

    for (int k0 = 0; k0 < K; k0 += 32) {
        __syncthreads();
        // stage A and B: each 128 rows x 32 halves = 512 b128 segments; 2 per thread
        #pragma unroll
        for (int i = 0; i < 2; ++i) {
            int idx = tid + i*256;
            int r = idx >> 2, s = idx & 3;
            *(v8h*)&sA[r*40 + s*8] =
                *(const v8h*)(A + (size_t)(mbase + r) * K + k0 + s*8);
            *(v8h*)&sB[r*40 + s*8] =
                *(const v8h*)(BT + (size_t)(nbase + r) * K + k0 + s*8);
        }
        __syncthreads();

        v16h a0 = lds_afrag(sA, wm*32,      lane);
        v16h a1 = lds_afrag(sA, wm*32 + 16, lane);
        v16h b0 = lds_bfrag(sB, wn*64,      lane);
        v16h b1 = lds_bfrag(sB, wn*64 + 16, lane);
        v16h b2 = lds_bfrag(sB, wn*64 + 32, lane);
        v16h b3 = lds_bfrag(sB, wn*64 + 48, lane);
        acc[0][0] = wmma16(a0, b0, acc[0][0]);
        acc[0][1] = wmma16(a0, b1, acc[0][1]);
        acc[0][2] = wmma16(a0, b2, acc[0][2]);
        acc[0][3] = wmma16(a0, b3, acc[0][3]);
        acc[1][0] = wmma16(a1, b0, acc[1][0]);
        acc[1][1] = wmma16(a1, b1, acc[1][1]);
        acc[1][2] = wmma16(a1, b2, acc[1][2]);
        acc[1][3] = wmma16(a1, b3, acc[1][3]);
    }

    float al = 0.0f;
    if (MODE == 1) al = alpha[0];

    #pragma unroll
    for (int i = 0; i < 2; ++i) {
        #pragma unroll
        for (int j = 0; j < 4; ++j) {
            int gc = nbase + wn*64 + j*16 + n_;
            float bv = bias[gc];
            #pragma unroll
            for (int r = 0; r < 8; ++r) {
                int gr = mbase + wm*32 + i*16 + hf*8 + r;
                float v = acc[i][j][r] + bv;
                if (MODE == 0) {
                    int which = gc >> 10, c = gc & 1023;
                    int hh = c >> 6, d = c & 63;
                    int b = gr >> 11, t = gr & 2047;
                    int bh = b*16 + hh;
                    if (which == 0)      qb  [((size_t)(bh*2048 + t))*64 + d]  = (_Float16)v;
                    else if (which == 1) kbuf[((size_t)(bh*2048 + t))*64 + d]  = (_Float16)v;
                    else                 vtb [((size_t)(bh*64 + d))*2048 + t]  = (_Float16)v;
                } else if (MODE == 1) {
                    size_t o = (size_t)gr * N + gc;
                    float xv = resid[o] + v;
                    outf[o] = xv;
                    outh[o] = (_Float16)(gamma[gc] * tanhf(xv * al) + beta[gc]);
                } else if (MODE == 2) {
                    size_t o = (size_t)gr * N + gc;
                    float g = 0.5f * v * (1.0f + erff(v * 0.70710678f));
                    outh[o] = (_Float16)g;
                } else {
                    size_t o = (size_t)gr * N + gc;
                    outf[o] = resid[o] + v;
                }
            }
        }
    }
}

// ============  flash attention (causal), one wave per 16-row Q tile  ============
// 64-key blocks: 4 score tiles + 8 PV WMMAs per block; softmax shuffles amortized
// over 64 keys. Only the last block needs causal masking (16-row q tile spans at
// most 64 trailing keys).
__global__ __launch_bounds__(128)
void attn_wmma(const _Float16* __restrict__ Q, const _Float16* __restrict__ Kb,
               const _Float16* __restrict__ VT, _Float16* __restrict__ O) {
    __shared__ __align__(16) _Float16 Pbuf[4][16 * 72];

    const int lane = threadIdx.x & 31;
    const int wl   = threadIdx.x >> 5;
    const int gw   = blockIdx.x * 4 + wl;
    const int bh   = gw >> 7;          // B*H = 32
    const int qt   = gw & 127;         // 128 q tiles of 16
    const int qbase = qt << 4;
    const int n_ = lane & 15, hf = lane >> 4;
    _Float16* Pw = Pbuf[wl];

    // Q fragments (d 0..31 and 32..63)
    const _Float16* qrow = Q + ((size_t)(bh*2048 + qbase + n_)) * 64;
    v16h qf0 = cat8(*(const v8h*)(qrow + hf*8),      *(const v8h*)(qrow + 16 + hf*8));
    v16h qf1 = cat8(*(const v8h*)(qrow + 32 + hf*8), *(const v8h*)(qrow + 48 + hf*8));

    float m[8], l[8];
    v8f o0 = {}, o1 = {}, o2 = {}, o3 = {};
    #pragma unroll
    for (int r = 0; r < 8; ++r) { m[r] = -3.0e38f; l[r] = 0.0f; }

    const int nkb = (qbase + 79) >> 6;   // 64-key blocks
    for (int kb = 0; kb < nkb; ++kb) {
        const int jb = kb << 6;

        // ---- scores: 4 tiles of 16 keys ----
        v8f s[4];
        #pragma unroll
        for (int t = 0; t < 4; ++t) {
            const _Float16* kp = Kb + ((size_t)(bh*2048 + jb + 16*t + n_)) * 64;
            v16h kc0 = cat8(*(const v8h*)(kp + hf*8),      *(const v8h*)(kp + hf*8 + 16));
            v16h kc1 = cat8(*(const v8h*)(kp + 32 + hf*8), *(const v8h*)(kp + 32 + hf*8 + 16));
            v8f z = {};
            s[t] = wmma16(qf0, kc0, z);
            s[t] = wmma16(qf1, kc1, s[t]);
        }

        const bool maskblk = (kb == nkb - 1);
        #pragma unroll
        for (int r = 0; r < 8; ++r) {
            const int qr = qbase + hf*8 + r;
            float a[4];
            #pragma unroll
            for (int t = 0; t < 4; ++t) {
                a[t] = s[t][r] * 0.125f;
                if (maskblk && (jb + 16*t + n_ > qr)) a[t] = -3.0e38f;
            }
            // row max across 16 lanes of this half
            float mr = fmaxf(fmaxf(a[0], a[1]), fmaxf(a[2], a[3]));
            mr = fmaxf(mr, __shfl_xor(mr, 1));
            mr = fmaxf(mr, __shfl_xor(mr, 2));
            mr = fmaxf(mr, __shfl_xor(mr, 4));
            mr = fmaxf(mr, __shfl_xor(mr, 8));
            float mn = fmaxf(m[r], mr);
            float corr = __expf(m[r] - mn);
            m[r] = mn;
            float e[4], rs = 0.0f;
            #pragma unroll
            for (int t = 0; t < 4; ++t) { e[t] = __expf(a[t] - mn); rs += e[t]; }
            rs += __shfl_xor(rs, 1);
            rs += __shfl_xor(rs, 2);
            rs += __shfl_xor(rs, 4);
            rs += __shfl_xor(rs, 8);
            l[r] = l[r] * corr + rs;
            o0[r] *= corr; o1[r] *= corr; o2[r] *= corr; o3[r] *= corr;
            #pragma unroll
            for (int t = 0; t < 4; ++t)
                Pw[(hf*8 + r)*72 + 16*t + n_] = (_Float16)e[t];
        }
        asm volatile("s_wait_dscnt 0" ::: "memory");

        // P as two A fragments (16x32 each over keys 0..31 / 32..63 of this block)
        v16h pf0 = cat8(*(const v8h*)(Pw + n_*72 + hf*8),
                        *(const v8h*)(Pw + n_*72 + 16 + hf*8));
        v16h pf1 = cat8(*(const v8h*)(Pw + n_*72 + 32 + hf*8),
                        *(const v8h*)(Pw + n_*72 + 48 + hf*8));

        // V fragments from transposed V [bh][d][t] (contiguous over keys)
        #pragma unroll
        for (int kb2 = 0; kb2 < 2; ++kb2) {
            const _Float16* vbase = VT + (size_t)(bh*64) * 2048 + jb + kb2*32 + hf*8;
            v16h vf0 = cat8(*(const v8h*)(vbase + (size_t)(0*16 + n_)*2048),
                            *(const v8h*)(vbase + (size_t)(0*16 + n_)*2048 + 16));
            v16h vf1 = cat8(*(const v8h*)(vbase + (size_t)(1*16 + n_)*2048),
                            *(const v8h*)(vbase + (size_t)(1*16 + n_)*2048 + 16));
            v16h vf2 = cat8(*(const v8h*)(vbase + (size_t)(2*16 + n_)*2048),
                            *(const v8h*)(vbase + (size_t)(2*16 + n_)*2048 + 16));
            v16h vf3 = cat8(*(const v8h*)(vbase + (size_t)(3*16 + n_)*2048),
                            *(const v8h*)(vbase + (size_t)(3*16 + n_)*2048 + 16));
            v16h pf = kb2 ? pf1 : pf0;
            o0 = wmma16(pf, vf0, o0);
            o1 = wmma16(pf, vf1, o1);
            o2 = wmma16(pf, vf2, o2);
            o3 = wmma16(pf, vf3, o3);
        }
    }

    const int b = bh >> 4, hh = bh & 15;
    #pragma unroll
    for (int r = 0; r < 8; ++r) {
        float inv = 1.0f / l[r];
        int trow = qbase + hf*8 + r;
        size_t orow = ((size_t)(b*2048 + trow)) * 1024 + hh*64;
        O[orow + 0*16 + n_] = (_Float16)(o0[r] * inv);
        O[orow + 1*16 + n_] = (_Float16)(o1[r] * inv);
        O[orow + 2*16 + n_] = (_Float16)(o2[r] * inv);
        O[orow + 3*16 + n_] = (_Float16)(o3[r] * inv);
    }
}

// ==========================================================================
extern "C" void kernel_launch(void* const* d_in, const int* in_sizes, int n_in,
                              void* d_out, int out_size, void* d_ws, size_t ws_size,
                              hipStream_t stream) {
    const float* x      = (const float*)d_in[0];
    const float* alpha  = (const float*)d_in[1];
    const float* gamma  = (const float*)d_in[2];
    const float* beta   = (const float*)d_in[3];
    const float* w_attn = (const float*)d_in[4];
    const float* b_attn = (const float*)d_in[5];
    const float* w_proj = (const float*)d_in[6];
    const float* b_proj = (const float*)d_in[7];
    const float* w_fc   = (const float*)d_in[8];
    const float* b_fc   = (const float*)d_in[9];
    const float* w_fc2  = (const float*)d_in[10];
    const float* b_fc2  = (const float*)d_in[11];
    float* out = (float*)d_out;

    char* ws = (char*)d_ws;
    size_t off = 0;
    auto alloc = [&](size_t bytes) { char* p = ws + off; off += (bytes + 255) & ~(size_t)255; return p; };

    _Float16* h     = (_Float16*)alloc((size_t)4096*1024*2);   // dyt(x) f16
    _Float16* waT   = (_Float16*)alloc((size_t)3072*1024*2);   // w_attn^T f16
    _Float16* wpT   = (_Float16*)alloc((size_t)1024*1024*2);   // w_proj^T f16
    _Float16* wfT   = (_Float16*)alloc((size_t)4096*1024*2);   // w_fc^T   f16
    _Float16* wf2T  = (_Float16*)alloc((size_t)1024*4096*2);   // w_fc2^T  f16
    _Float16* qbuf  = (_Float16*)alloc((size_t)32*2048*64*2);  // Q [bh][t][d]
    _Float16* kbuf  = (_Float16*)alloc((size_t)32*2048*64*2);  // K [bh][t][d]
    _Float16* vtb   = (_Float16*)alloc((size_t)32*64*2048*2);  // V [bh][d][t]
    _Float16* hattn = (_Float16*)alloc((size_t)4096*1024*2);   // attn out f16
    float*    x1    = (float*)   alloc((size_t)4096*1024*4);   // x + proj
    _Float16* h2    = (_Float16*)alloc((size_t)4096*1024*2);   // dyt(x1) f16
    _Float16* hg    = (_Float16*)alloc((size_t)4096*4096*2);   // gelu(fc) f16

    // 1) DyT -> f16
    dyt_f16_kernel<<<16384, 256, 0, stream>>>(x, alpha, gamma, beta, h);

    // 2) weight converts (f32 [K,N] -> f16 [N,K])
    cvt_transpose_kernel<<<(1024*3072)/256, 256, 0, stream>>>(w_attn, waT, 1024, 3072);
    cvt_transpose_kernel<<<(1024*1024)/256, 256, 0, stream>>>(w_proj, wpT, 1024, 1024);
    cvt_transpose_kernel<<<(1024*4096)/256, 256, 0, stream>>>(w_fc,   wfT, 1024, 4096);
    cvt_transpose_kernel<<<(4096*1024)/256, 256, 0, stream>>>(w_fc2,  wf2T, 4096, 1024);

    // 3) QKV GEMM: [4096,1024] x [1024,3072]
    gemm_wmma<0><<<dim3(3072/128, 4096/128), 256, 0, stream>>>(
        h, waT, b_attn, 4096, 3072, 1024,
        nullptr, nullptr, nullptr, qbuf, kbuf, vtb, nullptr, nullptr, nullptr);

    // 4) causal flash attention
    attn_wmma<<<1024, 128, 0, stream>>>(qbuf, kbuf, vtb, hattn);

    // 5) proj GEMM + residual + DyT
    gemm_wmma<1><<<dim3(1024/128, 4096/128), 256, 0, stream>>>(
        hattn, wpT, b_proj, 4096, 1024, 1024,
        x, x1, h2, nullptr, nullptr, nullptr, alpha, gamma, beta);

    // 6) FC GEMM + GELU
    gemm_wmma<2><<<dim3(4096/128, 4096/128), 256, 0, stream>>>(
        h2, wfT, b_fc, 4096, 4096, 1024,
        nullptr, nullptr, hg, nullptr, nullptr, nullptr, nullptr, nullptr, nullptr);

    // 7) FC2 GEMM + residual -> out (f32)
    gemm_wmma<3><<<dim3(1024/128, 4096/128), 256, 0, stream>>>(
        hg, wf2T, b_fc2, 4096, 1024, 4096,
        x1, out, nullptr, nullptr, nullptr, nullptr, nullptr, nullptr, nullptr);
}
